// gmm_hyper_y2_34565896798758
// MI455X (gfx1250) — matrix operand order repaired and checked by reading, not compiled, not run on hardware
//
#include <hip/hip_runtime.h>
#include <math.h>

typedef __attribute__((ext_vector_type(16))) _Float16 v16h;
typedef __attribute__((ext_vector_type(8)))  _Float16 v8h;
typedef __attribute__((ext_vector_type(8)))  float    v8f;

#define HW   96
#define PW   100                    // padded width: [-2, 97] halo for 5x5 pad=2
#define NB   8
#define NPIX (HW*HW)                // 9216
#define CCAT 320
#define CMID 128
#define CMK  576
#define SIGN ((size_t)NB*CMK*HW*HW) // 42467328 elems per big output

// ---- order-preserving float <-> uint for atomic max ----
__device__ __forceinline__ unsigned encf(float f) {
    unsigned u = __float_as_uint(f);
    return (u & 0x80000000u) ? ~u : (u | 0x80000000u);
}
__device__ __forceinline__ float decf(unsigned u) {
    unsigned v = (u & 0x80000000u) ? (u & 0x7FFFFFFFu) : ~u;
    return __uint_as_float(v);
}

// ---- upsample_bilinear_x4(z2) || y1 -> f16 x-padded NHWC act[(n*96+y)*100 + x+2][320]
__global__ void prep_act_kernel(const float* __restrict__ z2,
                                const float* __restrict__ y1,
                                _Float16* __restrict__ act) {
    const int total = NB * NPIX * CCAT;
    for (int idx = blockIdx.x * blockDim.x + threadIdx.x; idx < total;
         idx += gridDim.x * blockDim.x) {
        int c = idx % CCAT;
        int p = idx / CCAT;             // n*9216 + Y*96 + X
        int X = p % HW;
        int t = p / HW;
        int Y = t % HW;
        int n = t / HW;
        float v;
        if (c < 128) {
            float ys = Y * (23.0f / 95.0f);
            float xs = X * (23.0f / 95.0f);
            int y0 = (int)ys, x0 = (int)xs;
            int y1i = y0 + 1 < 23 ? y0 + 1 : 23;
            int x1i = x0 + 1 < 23 ? x0 + 1 : 23;
            float wy = ys - (float)y0, wx = xs - (float)x0;
            const float* zp = z2 + ((size_t)(n * 128 + c) * 24) * 24;
            float a00 = zp[y0 * 24 + x0],  a01 = zp[y0 * 24 + x1i];
            float a10 = zp[y1i * 24 + x0], a11 = zp[y1i * 24 + x1i];
            v = (a00 * (1.0f - wx) + a01 * wx) * (1.0f - wy) +
                (a10 * (1.0f - wx) + a11 * wx) * wy;
        } else {
            v = y1[((size_t)(n * 192 + (c - 128)) * HW + Y) * HW + X];
        }
        act[((size_t)(n * HW + Y) * PW + (X + 2)) * CCAT + c] = (_Float16)v;
    }
}

// ---- pack OIHW f32 weights into WMMA B-fragment layout, f16 ----
// Block (kykx, ciChunk, coTile) = 512 halves; element [lane*16 + h]:
//   co = coTile*16 + (lane&15),  ci = ciChunk*32 + 16*(h>>3) + 8*(lane>>4) + (h&7)
__global__ void pack_w_kernel(const float* __restrict__ w, _Float16* __restrict__ wp,
                              int Cin, int Cout) {
    const int nCC = Cin >> 5, nCT = Cout >> 4;
    const int total = 25 * Cin * Cout;
    for (int idx = blockIdx.x * blockDim.x + threadIdx.x; idx < total;
         idx += gridDim.x * blockDim.x) {
        int h    = idx & 15;
        int lane = (idx >> 4) & 31;
        int r    = idx >> 9;
        int ct = r % nCT;  r /= nCT;
        int cc = r % nCC;  int kk = r / nCC;
        int ky = kk / 5, kx = kk % 5;
        int co = ct * 16 + (lane & 15);
        int ci = cc * 32 + ((h >> 3) << 4) + ((lane >> 4) << 3) + (h & 7);
        wp[idx] = (_Float16)w[(((size_t)co * Cin + ci) * 5 + ky) * 5 + kx];
    }
}

__global__ void init_max_kernel(unsigned* mb) {
    int i = blockIdx.x * blockDim.x + threadIdx.x;
    if (i < NB * CMK) mb[i] = 0u;   // encf of most-negative
}

// ---- implicit-GEMM 5x5 conv, pad=2, via v_wmma_f32_16x16x32_f16 ----
// Wave tile: 16 pixels (one row segment) x 64 couts (4 accumulators; A fragment
// reused 4x, B fragments contiguous 4KB). Inputs are x-padded NHWC (stride PW)
// with zeroed halo -> A loads need NO predication. 8 waves/block = 8 pixel
// tiles; blockIdx.y = 64-wide cout tile. modes: 0 relu->f16 NHWC(padded),
// 1 lrelu->f16 NHWC(padded), 2 relu->f32 NCHW, 3 none->f32 NCHW, 4 max-reduce.
__global__ __launch_bounds__(256) void conv5_wmma_kernel(
    const _Float16* __restrict__ act, const _Float16* __restrict__ wp,
    const float* __restrict__ bias, int Cin, int Cout,
    _Float16* __restrict__ outH, float* __restrict__ outF,
    unsigned* __restrict__ maxbuf, int mode) {
    const int lane = threadIdx.x & 31;
    const int wave = threadIdx.x >> 5;
    const int g    = lane >> 4;
    const int m    = lane & 15;
    const int pixTile = blockIdx.x * 8 + wave;          // 0..4607
    const int ct0     = blockIdx.y * 4;                 // first of 4 co tiles
    const int pixBase = pixTile * 16;
    const int xb = pixBase % HW;
    int t = pixBase / HW;
    const int y = t % HW;
    const int n = t / HW;
    const int nCC = Cin >> 5;
    const int nCT = Cout >> 4;

    v8f acc[4];
    #pragma unroll
    for (int q = 0; q < 4; ++q)
        acc[q] = (v8f){0.f, 0.f, 0.f, 0.f, 0.f, 0.f, 0.f, 0.f};
    union V16 { v16h v; v8h h[2]; };

    // per-lane A base: pixel x = xb + m - 2 (kx added below), padded index +2
    const int px0 = xb + m;                             // padded x for kx=0
    for (int ky = 0; ky < 5; ++ky) {
        int sy = y + ky - 2;
        if (sy < 0 || sy > 95) continue;                // uniform per wave
        const _Float16* rowp =
            act + ((size_t)(n * HW + sy) * PW + px0) * Cin + g * 8;
        const _Float16* wrow =
            wp + ((size_t)(ky * 5) * nCC * nCT + ct0) * 512 + lane * 16;
        for (int kx = 0; kx < 5; ++kx) {
            const _Float16* ap0 = rowp + (size_t)kx * Cin;
            const _Float16* bp0 = wrow + (size_t)kx * nCC * nCT * 512;
            for (int cc = 0; cc < nCC; ++cc) {
                V16 a;
                a.h[0] = *(const v8h*)(ap0 + cc * 32);
                a.h[1] = *(const v8h*)(ap0 + cc * 32 + 16);
                const _Float16* bp = bp0 + (size_t)cc * nCT * 512;
                V16 b[4];
                #pragma unroll
                for (int q = 0; q < 4; ++q) {
                    b[q].h[0] = *(const v8h*)(bp + q * 512);
                    b[q].h[1] = *(const v8h*)(bp + q * 512 + 8);
                }
                #pragma unroll
                for (int q = 0; q < 4; ++q)
                    acc[q] = __builtin_amdgcn_wmma_f32_16x16x32_f16(
                        false, a.v, false, b[q].v, (short)0, acc[q], false, false);
            }
        }
    }

    if (mode == 4) {                                    // global max reduce
        #pragma unroll
        for (int q = 0; q < 4; ++q) {
            int co = (ct0 + q) * 16 + m;
            float bv = bias[co];
            float lm = -3.4e38f;
            #pragma unroll
            for (int r = 0; r < 8; ++r) lm = fmaxf(lm, acc[q][r] + bv);
            lm = fmaxf(lm, __shfl_xor(lm, 16, 32));     // lanes l / l+16 share co
            if (lane < 16) atomicMax(&maxbuf[n * CMK + co], encf(lm));
        }
        return;
    }
    #pragma unroll
    for (int q = 0; q < 4; ++q) {
        int co = (ct0 + q) * 16 + m;
        float bv = bias[co];
        #pragma unroll
        for (int r = 0; r < 8; ++r) {
            float v = acc[q][r] + bv;
            int mo = r + 8 * g;                         // pixel in tile (C layout)
            int x  = xb + mo;
            if (mode == 0) {
                v = v > 0.f ? v : 0.f;
                outH[((size_t)(n * HW + y) * PW + (x + 2)) * Cout + co] = (_Float16)v;
            } else if (mode == 1) {
                v = v >= 0.f ? v : 0.01f * v;
                outH[((size_t)(n * HW + y) * PW + (x + 2)) * Cout + co] = (_Float16)v;
            } else if (mode == 2) {
                v = v > 0.f ? v : 0.f;
                outF[(((size_t)n * Cout + co) * HW + y) * HW + x] = v;
            } else {
                outF[(((size_t)n * Cout + co) * HW + y) * HW + x] = v;
            }
        }
    }
}

// ---- weights branch tail: lrelu -> 1x1 conv (576x576) -> softmax over K=3 ----
__global__ void weights_tail_kernel(const unsigned* __restrict__ mb,
                                    const float* __restrict__ ww3,
                                    const float* __restrict__ bw3,
                                    float* __restrict__ out) {
    __shared__ float sh[CMK];
    __shared__ float lg[CMK];
    const int n  = blockIdx.x;
    const int co = threadIdx.x;
    float hv = decf(mb[n * CMK + co]);
    hv = hv >= 0.f ? hv : 0.01f * hv;
    sh[co] = hv;
    __syncthreads();
    float acc = bw3[co];
    for (int ci = 0; ci < CMK; ++ci) acc += ww3[(size_t)co * CMK + ci] * sh[ci];
    lg[co] = acc;
    __syncthreads();
    int mc = co % 192;
    float a0 = lg[mc], a1 = lg[192 + mc], a2 = lg[384 + mc];
    float mx = fmaxf(a0, fmaxf(a1, a2));
    float den = expf(a0 - mx) + expf(a1 - mx) + expf(a2 - mx);
    out[n * CMK + co] = expf(acc - mx) / den;
}

extern "C" void kernel_launch(void* const* d_in, const int* in_sizes, int n_in,
                              void* d_out, int out_size, void* d_ws, size_t ws_size,
                              hipStream_t stream) {
    (void)in_sizes; (void)n_in; (void)out_size; (void)ws_size;
    const float* z2  = (const float*)d_in[0];
    const float* y1  = (const float*)d_in[1];
    const float* ws1 = (const float*)d_in[2];  const float* bs1 = (const float*)d_in[3];
    const float* ws2 = (const float*)d_in[4];  const float* bs2 = (const float*)d_in[5];
    const float* ws3 = (const float*)d_in[6];  const float* bs3 = (const float*)d_in[7];
    const float* wm1 = (const float*)d_in[8];  const float* bm1 = (const float*)d_in[9];
    const float* wm2 = (const float*)d_in[10]; const float* bm2 = (const float*)d_in[11];
    const float* wm3 = (const float*)d_in[12]; const float* bm3 = (const float*)d_in[13];
    const float* ww1 = (const float*)d_in[14]; const float* bw1 = (const float*)d_in[15];
    const float* ww2 = (const float*)d_in[16]; const float* bw2 = (const float*)d_in[17];
    const float* ww3 = (const float*)d_in[18]; const float* bw3 = (const float*)d_in[19];
    float* out = (float*)d_out;

    char* p = (char*)d_ws;
    auto alloc = [&](size_t bytes) -> char* {
        char* r = p;
        p += (bytes + 255) & ~(size_t)255;
        return r;
    };
    const size_t actB  = (size_t)NB * HW * PW * CCAT * 2;   // x-padded
    const size_t bufBy = (size_t)NB * HW * PW * CMID * 2;   // x-padded
    _Float16* act0 = (_Float16*)alloc(actB);
    _Float16* bufA = (_Float16*)alloc(bufBy);
    _Float16* bufB = (_Float16*)alloc(bufBy);
    _Float16* wp1s = (_Float16*)alloc((size_t)25 * CCAT * CMID * 2);
    _Float16* wp2s = (_Float16*)alloc((size_t)25 * CMID * CMID * 2);
    _Float16* wp3s = (_Float16*)alloc((size_t)25 * CMID * CMK * 2);
    _Float16* wp1m = (_Float16*)alloc((size_t)25 * CCAT * CMID * 2);
    _Float16* wp2m = (_Float16*)alloc((size_t)25 * CMID * CMID * 2);
    _Float16* wp3m = (_Float16*)alloc((size_t)25 * CMID * CMK * 2);
    _Float16* wp1w = (_Float16*)alloc((size_t)25 * CCAT * CMID * 2);
    _Float16* wp2w = (_Float16*)alloc((size_t)25 * CMID * CMK * 2);
    unsigned* maxb = (unsigned*)alloc((size_t)NB * CMK * 4);

    // zero padded activation buffers (halo must be 0; interior overwritten)
    hipMemsetAsync(act0, 0, actB, stream);
    hipMemsetAsync(bufA, 0, bufBy, stream);
    hipMemsetAsync(bufB, 0, bufBy, stream);

    // weight packing (f32 OIHW -> f16 WMMA B-fragment blocks)
    int tBig = 25 * CCAT * CMID, tMid = 25 * CMID * CMID, tOut = 25 * CMID * CMK;
    pack_w_kernel<<<(tBig + 255) / 256, 256, 0, stream>>>(ws1, wp1s, CCAT, CMID);
    pack_w_kernel<<<(tMid + 255) / 256, 256, 0, stream>>>(ws2, wp2s, CMID, CMID);
    pack_w_kernel<<<(tOut + 255) / 256, 256, 0, stream>>>(ws3, wp3s, CMID, CMK);
    pack_w_kernel<<<(tBig + 255) / 256, 256, 0, stream>>>(wm1, wp1m, CCAT, CMID);
    pack_w_kernel<<<(tMid + 255) / 256, 256, 0, stream>>>(wm2, wp2m, CMID, CMID);
    pack_w_kernel<<<(tOut + 255) / 256, 256, 0, stream>>>(wm3, wp3m, CMID, CMK);
    pack_w_kernel<<<(tBig + 255) / 256, 256, 0, stream>>>(ww1, wp1w, CCAT, CMID);
    pack_w_kernel<<<(tOut + 255) / 256, 256, 0, stream>>>(ww2, wp2w, CMID, CMK);

    // upsample + concat into padded f16 NHWC
    int tAct = NB * NPIX * CCAT;
    prep_act_kernel<<<(tAct + 255) / 256, 256, 0, stream>>>(z2, y1, act0);
    init_max_kernel<<<(NB * CMK + 255) / 256, 256, 0, stream>>>(maxb);

    dim3 blk(256);
    dim3 gMid(NB * NPIX / 16 / 8, CMID / 64);   // 576 x 2
    dim3 gOut(NB * NPIX / 16 / 8, CMK / 64);    // 576 x 9

    // sigma branch
    conv5_wmma_kernel<<<gMid, blk, 0, stream>>>(act0, wp1s, bs1, CCAT, CMID, bufA, nullptr, nullptr, 0);
    conv5_wmma_kernel<<<gMid, blk, 0, stream>>>(bufA, wp2s, bs2, CMID, CMID, bufB, nullptr, nullptr, 0);
    conv5_wmma_kernel<<<gOut, blk, 0, stream>>>(bufB, wp3s, bs3, CMID, CMK, nullptr, out, nullptr, 2);
    // means branch
    conv5_wmma_kernel<<<gMid, blk, 0, stream>>>(act0, wp1m, bm1, CCAT, CMID, bufA, nullptr, nullptr, 1);
    conv5_wmma_kernel<<<gMid, blk, 0, stream>>>(bufA, wp2m, bm2, CMID, CMID, bufB, nullptr, nullptr, 1);
    conv5_wmma_kernel<<<gOut, blk, 0, stream>>>(bufB, wp3m, bm3, CMID, CMK, nullptr, out + SIGN, nullptr, 3);
    // weights branch
    conv5_wmma_kernel<<<gMid, blk, 0, stream>>>(act0, wp1w, bw1, CCAT, CMID, bufA, nullptr, nullptr, 1);
    conv5_wmma_kernel<<<gOut, blk, 0, stream>>>(bufA, wp2w, bw2, CMID, CMK, nullptr, nullptr, maxb, 4);
    weights_tail_kernel<<<NB, CMK, 0, stream>>>(maxb, ww3, bw3, out + 2 * SIGN);
}
